// EGCN_83648783057032
// MI455X (gfx1250) — matrix operand order repaired and checked by reading, not compiled, not run on hardware
//
#include <hip/hip_runtime.h>

// ---------------------------------------------------------------------------
// EvolveGCN (T=8, N=4096, F=256) for gfx1250 — bf16 WMMA + TDM async staging.
// Only h2[-1] is needed, so layer-1 heavy GEMMs run once, and layer-0's big
// A@Y GEMM is full-M only at t=7 (t<7 need only the first 256 output rows).
// ---------------------------------------------------------------------------

typedef __bf16 bf16_t;
typedef __attribute__((ext_vector_type(16))) __bf16 v16bf;
typedef __attribute__((ext_vector_type(8)))  float  v8f;
typedef __attribute__((ext_vector_type(4)))  float  v4f;
typedef __attribute__((ext_vector_type(4)))  unsigned int u32x4;
typedef __attribute__((ext_vector_type(8)))  int    i32x8;
typedef __attribute__((ext_vector_type(4)))  int    i32x4;

#define FDIM 256
#define NN   4096
#define TT   8
#define NTILES 16       // 256/16 output column tiles per wave strip
#define BROW 40         // bf16 elements per padded LDS B-row (80 B: 20-bank stride)

// --- A-matrix fragment (16x32 bf16, M x K) built from f32 global memory ----
// Layout: lane<16 -> row M=lane,   K = k0+0..15
//         lane>=16 -> row M=lane-16, K = k0+16..31
__device__ __forceinline__ v16bf load_a_frag_f32(const float* __restrict__ A,
                                                 int lda, int mrow, int k0,
                                                 int lane) {
  int base = (lane & 16) ? 16 : 0;
  const float* p = A + (size_t)mrow * lda + k0 + base;
  v4f f0 = *(const v4f*)(p + 0);
  v4f f1 = *(const v4f*)(p + 4);
  v4f f2 = *(const v4f*)(p + 8);
  v4f f3 = *(const v4f*)(p + 12);
  v16bf r;
#pragma unroll
  for (int i = 0; i < 4; ++i) {
    r[i]      = (bf16_t)f0[i];
    r[4 + i]  = (bf16_t)f1[i];
    r[8 + i]  = (bf16_t)f2[i];
    r[12 + i] = (bf16_t)f3[i];
  }
  return r;
}

// --- B-matrix fragment (32x16, K x N) from a column-contiguous bf16 store ---
// Bt is [N][K] bf16 row-major (B stored transposed): lane<16 -> col N=lane,
// K=k0+0..15 ; lane>=16 -> col N=lane-16, K=k0+16..31. One 32-byte load.
__device__ __forceinline__ v16bf load_bt_frag(const bf16_t* __restrict__ Bt,
                                              int ldb, int n0, int k0,
                                              int lane) {
  int n  = n0 + (lane & 15);
  int ks = k0 + ((lane & 16) ? 16 : 0);
  return *(const v16bf*)(Bt + (size_t)n * ldb + ks);
}

__device__ __forceinline__ v8f wmma_bf16(v16bf a, v16bf b, v8f c) {
  return __builtin_amdgcn_wmma_f32_16x16x32_bf16(false, a, false, b,
                                                 (short)0, c, false, false);
}

// ---------------------------------------------------------------------------
// TDM: async-load one 32(K) x 256(N) bf16 panel (row stride = K elements)
// into LDS with 80-byte padded rows (pad_interval=16 DW, pad_amount=4 DW).
// Issued once per wave (wave-level op, descriptor in SGPRs). TENSORcnt-tracked.
// ---------------------------------------------------------------------------
__device__ __forceinline__ void tdm_load_panel(const bf16_t* gsrc,
                                               unsigned lds_off, int K) {
  unsigned long long ga = (unsigned long long)(uintptr_t)gsrc;
  u32x4 g0;
  g0[0] = 1u;                                        // count=1, user mode
  g0[1] = lds_off;                                   // lds_addr [63:32]
  g0[2] = (unsigned)(ga & 0xFFFFFFFFu);              // global_addr lo
  g0[3] = (unsigned)((ga >> 32) & 0x1FFFFFFu)        // global_addr hi (57b)
          | (2u << 30);                              // type=2 (image/tensor)
  i32x8 g1;
  g1[0] = (1 << 16)                                  // data_size = 2 bytes
          | (1 << 20)                                // pad_enable
          | (3 << 22)                                // pad_interval: 16 DWORDs
          | (3 << 25);                               // pad_amount: 4 DWORDs
  g1[1] = (32 << 16);                                // tensor_dim0 = 32 (lo16)
  g1[2] = (256 << 16);                               // tensor_dim1 = 256 (lo16)
  g1[3] = (32 << 16);                                // tile_dim0 = 32
  g1[4] = 256;                                       // tile_dim1 = 256
  g1[5] = K;                                         // tensor_dim0_stride lo32
  g1[6] = 0;
  g1[7] = 0;
  i32x4 z4 = {0, 0, 0, 0};
#if __has_include(<hip/amd_detail/amd_gfx1250_TDM.h>)
  i32x8 z8 = {0, 0, 0, 0, 0, 0, 0, 0};
  __builtin_amdgcn_tensor_load_to_lds(g0, g1, z4, z4, z8, 0);
#else
  __builtin_amdgcn_tensor_load_to_lds(g0, g1, z4, z4, 0);
#endif
}

// load two B fragments (n-tiles 2g, 2g+1) from the LDS panel
__device__ __forceinline__ void load_b2(const bf16_t* bp, int g, int ncol,
                                        int ksel, v16bf* b) {
#pragma unroll
  for (int i = 0; i < 2; ++i)
    b[i] = *(const v16bf*)(bp + ((g * 2 + i) * 16 + ncol) * BROW + ksel);
}

// ---------------------------------------------------------------------------
// Unified GEMM: C[M,256] = A[M,K](f32) x B[K,256], B given transposed (bf16).
// 8 waves/block, wave = 16-row strip x full 256 cols (16 accum tiles).
// B panels staged in LDS by the Tensor Data Mover, double-buffered. B
// fragments use two disjoint register buffers scheduled two groups ahead
// (no copy chain -> allocator cannot re-serialize the ds_load/WMMA overlap);
// A fragments pipelined one k-chunk ahead from global.
// Optional f32 output (with relu) and/or bf16 transposed output (for Y^T).
// ---------------------------------------------------------------------------
__global__ __launch_bounds__(256) void gemm_bf16_n256(
    const float* __restrict__ Abase, long long strideA, int lda,
    const bf16_t* __restrict__ Btbase, long long strideB,
    float* __restrict__ Cf, long long strideCf, int ldc,
    bf16_t* __restrict__ Ct, long long strideCt, int ldct,
    int K, int relu) {
  __shared__ __align__(128) bf16_t bpan[2][FDIM * BROW];  // 2 x 20 KB

  int t = blockIdx.y;
  const float*  A  = Abase  + (size_t)t * strideA;
  const bf16_t* Bt = Btbase + (size_t)t * strideB;

  int lane = threadIdx.x & 31;
  int wave = threadIdx.x >> 5;
  int tm   = (blockIdx.x * 8 + wave) * 16;
  int mrow = tm + (lane & 15);
  int ncol = lane & 15;
  int ksel = (lane & 16) ? 16 : 0;

  v8f acc[NTILES];
#pragma unroll
  for (int i = 0; i < NTILES; ++i)
#pragma unroll
    for (int r = 0; r < 8; ++r) acc[i][r] = 0.0f;

  if (wave == 0)  // prologue: DMA panel 0
    tdm_load_panel(Bt, (unsigned)(uintptr_t)&bpan[0][0], K);
  v16bf a_cur = load_a_frag_f32(A, lda, mrow, 0, lane);  // prologue A frag
  v16bf a_next = a_cur;

  for (int k0 = 0; k0 < K; k0 += 32) {
    int cur = (k0 >> 5) & 1;
    if (wave == 0) {
      if (k0 + 32 < K) {  // DMA next panel, retire current
        tdm_load_panel(Bt + (k0 + 32),
                       (unsigned)(uintptr_t)&bpan[cur ^ 1][0], K);
        __builtin_amdgcn_s_wait_tensorcnt(1);
      } else {
        __builtin_amdgcn_s_wait_tensorcnt(0);
      }
    }
    __syncthreads();  // panel `cur` visible to all waves

    // issue next A fragment early; its latency hides under 16 WMMAs
    if (k0 + 32 < K)
      a_next = load_a_frag_f32(A, lda, mrow, k0 + 32, lane);

    const bf16_t* bp = &bpan[cur][0];
    // 8 groups of 2 n-tiles; group g's fragments loaded 2 groups ahead in
    // alternating disjoint buffers b0/b1 (no register copy chain).
    v16bf b0[2], b1[2];
    load_b2(bp, 0, ncol, ksel, b0);
    load_b2(bp, 1, ncol, ksel, b1);
#pragma unroll
    for (int g = 0; g < 8; ++g) {
      v16bf* bc = (g & 1) ? b1 : b0;
      acc[2 * g]     = wmma_bf16(a_cur, bc[0], acc[2 * g]);
      acc[2 * g + 1] = wmma_bf16(a_cur, bc[1], acc[2 * g + 1]);
      if (g + 2 < 8)  // refill the buffer just consumed, 2 groups ahead
        load_b2(bp, g + 2, ncol, ksel, bc);
    }
    a_cur = a_next;
    __syncthreads();  // compute done before buffer `cur` is overwritten
  }

  // C layout: vgpr r, lane l -> M = tm + r + 8*(l>=16), N = ntile*16 + (l&15)
  int mbase = tm + ((lane & 16) ? 8 : 0);
#pragma unroll
  for (int nt = 0; nt < NTILES; ++nt) {
    int n = nt * 16 + ncol;
#pragma unroll
    for (int r = 0; r < 8; ++r) {
      float v = acc[nt][r];
      if (relu) v = v > 0.0f ? v : 0.0f;
      int m = mbase + r;
      if (Cf) Cf[(size_t)t * strideCf + (size_t)m * ldc + n] = v;
      if (Ct) Ct[(size_t)t * strideCt + (size_t)n * ldct + m] = (bf16_t)v;
    }
  }
}

// ---------------------------------------------------------------------------
// TopK/zt: score_j = (embs_j . scorer)/||scorer||; ztb[t][j][f] = e*tanh(score)
// ztb rows are exactly zt columns -> ready-made transposed-B operand.
// ---------------------------------------------------------------------------
__global__ void topk_ztb(const float* __restrict__ Ebase, long long strideE,
                         const float* __restrict__ Elast,
                         const float* __restrict__ scorer,
                         bf16_t* __restrict__ ZtBall) {
  int t = blockIdx.y, j = blockIdx.x, f = threadIdx.x;
  const float* E = (Elast != nullptr && t == TT - 1)
                       ? Elast
                       : (Ebase + (size_t)t * strideE);
  __shared__ float red[FDIM];
  __shared__ float red2[FDIM];
  float sv = scorer[f];
  float ev = E[(size_t)j * FDIM + f];
  red[f]  = ev * sv;
  red2[f] = sv * sv;
  __syncthreads();
  for (int s = FDIM / 2; s > 0; s >>= 1) {
    if (f < s) { red[f] += red[f + s]; red2[f] += red2[f + s]; }
    __syncthreads();
  }
  float th = tanhf(red[0] / sqrtf(red2[0]));
  ZtBall[((size_t)t * FDIM + j) * FDIM + f] = (bf16_t)(ev * th);
}

// Q state init: Qcur (f32) and Qcur^T (bf16)
__global__ void q_init(const float* __restrict__ Q0, float* __restrict__ QcurF,
                       bf16_t* __restrict__ QcurT) {
  int m = blockIdx.x, n = threadIdx.x;
  float q = Q0[(size_t)m * FDIM + n];
  QcurF[(size_t)m * FDIM + n] = q;
  QcurT[(size_t)n * FDIM + m] = (bf16_t)q;
}

// gate = sigmoid(W @ zt + U @ Q + b), one wave per 16x256 strip
__global__ __launch_bounds__(32) void gru_gate(
    const float* __restrict__ W, const float* __restrict__ U,
    const float* __restrict__ bias, const bf16_t* __restrict__ ZtB,
    const bf16_t* __restrict__ QT, float* __restrict__ outgate) {
  int lane = threadIdx.x & 31;
  int tm   = blockIdx.x * 16;
  int mrow = tm + (lane & 15);

  v8f acc[NTILES];
#pragma unroll
  for (int i = 0; i < NTILES; ++i)
#pragma unroll
    for (int r = 0; r < 8; ++r) acc[i][r] = 0.0f;

  v16bf aw = load_a_frag_f32(W, FDIM, mrow, 0, lane);
  v16bf au = load_a_frag_f32(U, FDIM, mrow, 0, lane);
  for (int k0 = 0; k0 < FDIM; k0 += 32) {
    v16bf aw_n = aw, au_n = au;
    if (k0 + 32 < FDIM) {  // pipeline next k-chunk A fragments
      aw_n = load_a_frag_f32(W, FDIM, mrow, k0 + 32, lane);
      au_n = load_a_frag_f32(U, FDIM, mrow, k0 + 32, lane);
    }
    v16bf bz = load_bt_frag(ZtB, FDIM, 0, k0, lane);
    v16bf bq = load_bt_frag(QT, FDIM, 0, k0, lane);
#pragma unroll
    for (int nt = 0; nt < NTILES; ++nt) {
      v16bf bz_n = bz, bq_n = bq;
      if (nt + 1 < NTILES) {  // pipeline next n-tile B fragments
        bz_n = load_bt_frag(ZtB, FDIM, (nt + 1) * 16, k0, lane);
        bq_n = load_bt_frag(QT, FDIM, (nt + 1) * 16, k0, lane);
      }
      acc[nt] = wmma_bf16(aw, bz, acc[nt]);
      acc[nt] = wmma_bf16(au, bq, acc[nt]);
      bz = bz_n; bq = bq_n;
    }
    aw = aw_n; au = au_n;
  }

  int mbase = tm + ((lane & 16) ? 8 : 0);
  int ncol  = (lane & 15);
#pragma unroll
  for (int nt = 0; nt < NTILES; ++nt) {
    int n = nt * 16 + ncol;
#pragma unroll
    for (int r = 0; r < 8; ++r) {
      int m = mbase + r;
      float x = acc[nt][r] + bias[(size_t)m * FDIM + n];
      outgate[(size_t)m * FDIM + n] = 1.0f / (1.0f + __expf(-x));
    }
  }
}

// RQT[n][m] = bf16(rst[m][n] * Q[m][n])  (transposed-B operand for Uh GEMM)
__global__ void rqt_kernel(const float* __restrict__ rst,
                           const float* __restrict__ QcurF,
                           bf16_t* __restrict__ RQT) {
  int m = blockIdx.x, n = threadIdx.x;
  size_t i = (size_t)m * FDIM + n;
  RQT[(size_t)n * FDIM + m] = (bf16_t)(rst[i] * QcurF[i]);
}

// hcap = tanh(Wh@zt + Uh@(rst*Q) + bh); Qn = (1-u)*Q + u*hcap (in-place state)
__global__ __launch_bounds__(32) void gru_hq(
    const float* __restrict__ Wh, const float* __restrict__ Uh,
    const float* __restrict__ bh, const bf16_t* __restrict__ ZtB,
    const bf16_t* __restrict__ RQT, const float* __restrict__ upd,
    float* __restrict__ QcurF, bf16_t* __restrict__ QcurT,
    bf16_t* __restrict__ QTs_t) {
  int lane = threadIdx.x & 31;
  int tm   = blockIdx.x * 16;
  int mrow = tm + (lane & 15);

  v8f acc[NTILES];
#pragma unroll
  for (int i = 0; i < NTILES; ++i)
#pragma unroll
    for (int r = 0; r < 8; ++r) acc[i][r] = 0.0f;

  v16bf aw = load_a_frag_f32(Wh, FDIM, mrow, 0, lane);
  v16bf au = load_a_frag_f32(Uh, FDIM, mrow, 0, lane);
  for (int k0 = 0; k0 < FDIM; k0 += 32) {
    v16bf aw_n = aw, au_n = au;
    if (k0 + 32 < FDIM) {
      aw_n = load_a_frag_f32(Wh, FDIM, mrow, k0 + 32, lane);
      au_n = load_a_frag_f32(Uh, FDIM, mrow, k0 + 32, lane);
    }
    v16bf bz = load_bt_frag(ZtB, FDIM, 0, k0, lane);
    v16bf bq = load_bt_frag(RQT, FDIM, 0, k0, lane);
#pragma unroll
    for (int nt = 0; nt < NTILES; ++nt) {
      v16bf bz_n = bz, bq_n = bq;
      if (nt + 1 < NTILES) {
        bz_n = load_bt_frag(ZtB, FDIM, (nt + 1) * 16, k0, lane);
        bq_n = load_bt_frag(RQT, FDIM, (nt + 1) * 16, k0, lane);
      }
      acc[nt] = wmma_bf16(aw, bz, acc[nt]);
      acc[nt] = wmma_bf16(au, bq, acc[nt]);
      bz = bz_n; bq = bq_n;
    }
    aw = aw_n; au = au_n;
  }

  int mbase = tm + ((lane & 16) ? 8 : 0);
  int ncol  = (lane & 15);
#pragma unroll
  for (int nt = 0; nt < NTILES; ++nt) {
    int n = nt * 16 + ncol;
#pragma unroll
    for (int r = 0; r < 8; ++r) {
      int m = mbase + r;
      size_t i = (size_t)m * FDIM + n;
      float h = tanhf(acc[nt][r] + bh[i]);
      float u = upd[i];
      float q = QcurF[i];
      float qn = (1.0f - u) * q + u * h;
      QcurF[i] = qn;                                   // next-step state
      bf16_t qb = (bf16_t)qn;
      QcurT[(size_t)n * FDIM + m] = qb;                // transposed state
      QTs_t[(size_t)n * FDIM + m] = qb;                // history for GEMM1
    }
  }
}

// ---------------------------------------------------------------------------
extern "C" void kernel_launch(void* const* d_in, const int* in_sizes, int n_in,
                              void* d_out, int out_size, void* d_ws,
                              size_t ws_size, hipStream_t stream) {
  (void)in_sizes; (void)n_in; (void)out_size; (void)ws_size;
  const float* Ain = (const float*)d_in[0];  // [T,N,N]
  const float* X   = (const float*)d_in[1];  // [T,N,F]
  auto w = [&](int i) { return (const float*)d_in[i]; };
  // per-layer weight index bases: l0 -> 3..13, l1 -> 14..24
  // order: scorer, Wz,Uz,bz, Wr,Ur,br, Wh,Uh,bh, Q0

  char* ws = (char*)d_ws;
  size_t off = 0;
  auto alloc = [&](size_t bytes) {
    off = (off + 255) & ~(size_t)255;
    void* p = ws + off;
    off += bytes;
    return p;
  };
  const size_t FF = (size_t)FDIM * FDIM;               // 65536
  bf16_t* ZtB    = (bf16_t*)alloc(TT * FF * 2);        // [T,256,256] bf16
  float*  QcurF  = (float*) alloc(FF * 4);
  bf16_t* QcurT  = (bf16_t*)alloc(FF * 2);
  bf16_t* QTs    = (bf16_t*)alloc(TT * FF * 2);        // Qn^T history
  float*  updF   = (float*) alloc(FF * 4);
  float*  rstF   = (float*) alloc(FF * 4);
  bf16_t* RQT    = (bf16_t*)alloc(FF * 2);
  bf16_t* Yt     = (bf16_t*)alloc((size_t)TT * FDIM * NN * 2); // Y^T [T,256,4096]
  float*  h1s    = (float*) alloc(TT * FF * 4);        // h1[t][:256] rows
  float*  h1last = (float*) alloc((size_t)NN * FDIM * 4);

  const long long Ystride = (long long)FDIM * NN;      // 256*4096
  const long long Astride = (long long)NN * NN;

  auto run_gru_chain = [&](int base, const bf16_t* ztb_all) {
    q_init<<<FDIM, FDIM, 0, stream>>>(w(base + 10), QcurF, QcurT);
    for (int t = 0; t < TT; ++t) {
      const bf16_t* ztb = ztb_all + (size_t)t * FF;
      gru_gate<<<16, 32, 0, stream>>>(w(base + 1), w(base + 2), w(base + 3),
                                      ztb, QcurT, updF);
      gru_gate<<<16, 32, 0, stream>>>(w(base + 4), w(base + 5), w(base + 6),
                                      ztb, QcurT, rstF);
      rqt_kernel<<<FDIM, FDIM, 0, stream>>>(rstF, QcurF, RQT);
      gru_hq<<<16, 32, 0, stream>>>(w(base + 7), w(base + 8), w(base + 9), ztb,
                                    RQT, updF, QcurF, QcurT, QTs + (size_t)t * FF);
    }
  };

  // ----- layer 0 -----
  topk_ztb<<<dim3(FDIM, TT), FDIM, 0, stream>>>(X, (long long)NN * FDIM,
                                                nullptr, w(3), ZtB);
  run_gru_chain(3, ZtB);
  // GEMM1 (all t): Y[t]^T = (X[t] @ Qn[t])^T  -> bf16 transposed out
  gemm_bf16_n256<<<dim3(NN / 128, TT), 256, 0, stream>>>(
      X, (long long)NN * FDIM, FDIM, QTs, (long long)FF, nullptr, 0, 0, Yt,
      Ystride, NN, FDIM, 0);
  // GEMM2 t<7: only first 256 output rows are ever consumed
  gemm_bf16_n256<<<dim3(2, TT - 1), 256, 0, stream>>>(
      Ain, Astride, NN, Yt, Ystride, h1s, (long long)FF, FDIM, nullptr, 0, 0,
      NN, 1);
  // GEMM2 t=7: full M=4096 (feeds layer-1 GEMM1)
  gemm_bf16_n256<<<dim3(NN / 128, 1), 256, 0, stream>>>(
      Ain + (size_t)(TT - 1) * NN * NN, 0, NN, Yt + (size_t)(TT - 1) * Ystride,
      0, h1last, 0, FDIM, nullptr, 0, 0, NN, 1);

  // ----- layer 1 -----
  topk_ztb<<<dim3(FDIM, TT), FDIM, 0, stream>>>(h1s, (long long)FF, h1last,
                                                w(14), ZtB);
  run_gru_chain(14, ZtB);
  // GEMM1 only t=7: Y^T = (h1[7] @ Qn1[7])^T
  gemm_bf16_n256<<<dim3(NN / 128, 1), 256, 0, stream>>>(
      h1last, 0, FDIM, QTs + (size_t)(TT - 1) * FF, 0, nullptr, 0, 0, Yt, 0,
      NN, FDIM, 0);
  // final: out = relu(A[7] @ Y) -> d_out [4096,256] f32
  gemm_bf16_n256<<<dim3(NN / 128, 1), 256, 0, stream>>>(
      Ain + (size_t)(TT - 1) * NN * NN, 0, NN, Yt, 0, (float*)d_out, 0, FDIM,
      nullptr, 0, 0, NN, 1);
}
